// QuantumCouplingLayer_3358664425848
// MI455X (gfx1250) — matrix operand order repaired and checked by reading, not compiled, not run on hardware
//
#include <hip/hip_runtime.h>
#include <hip/hip_bf16.h>
#include <math.h>

// Problem constants
#define BB 8
#define SS 1024
#define LLIMB 8
#define DD 512
#define NTOK (BB*SS)            // 8192 tokens
#define NROW (NTOK*LLIMB)       // 65536 rows
#define NOUT ((size_t)NROW*DD)  // 33554432 output elements (then 3 scalars)

typedef __bf16 bf16_t;
typedef __bf16 v16bf __attribute__((ext_vector_type(16)));
typedef float  v8f   __attribute__((ext_vector_type(8)));

union FragBF { v16bf v; uint4 q[2]; };

static __device__ __forceinline__ v8f wmma_bf16(v16bf a, v16bf b, v8f c) {
  // D = A(16x32 bf16) * B(32x16 bf16) + C(16x16 f32)
  return __builtin_amdgcn_wmma_f32_16x16x32_bf16(false, a, false, b, (short)0, c, false, false);
}

// ---------------------------------------------------------------------------
// Prep: symmetric zero-diag coupling matrix + Rabi amplitudes
// ---------------------------------------------------------------------------
__global__ void prep_small(const float* __restrict__ c,
                           float* __restrict__ Cs, float* __restrict__ ampS) {
  int i = threadIdx.x;
  if (i < 64) {
    int r = i >> 3, cc = i & 7;
    float v = 0.5f * (c[r*8+cc] + c[cc*8+r]);
    if (r == cc) v = 0.f;
    Cs[i] = v;
  }
  if (i < 8) {
    int j = (i + 1) & 7;                    // ring neighbor, never diagonal
    float v = 0.5f * (c[i*8+j] + c[j*8+i]);
    float s = sinf(fabsf(v) * 1.0f);        // TIME_STEP = 1.0
    ampS[i] = s * s;
  }
}

// Convert weights to bf16 (Wc: 512x512, Wg: 512x1024)
__global__ void cvt_weights(const float* __restrict__ Wc, const float* __restrict__ Wg,
                            bf16_t* __restrict__ WcB, bf16_t* __restrict__ WgB) {
  int i = blockIdx.x * 256 + threadIdx.x;   // grid covers 512*1024
  if (i < 512*512)  WcB[i] = (bf16_t)Wc[i];
  if (i < 512*1024) WgB[i] = (bf16_t)Wg[i];
}

// ---------------------------------------------------------------------------
// Fused main kernel: proj GEMM -> coupling/exchange -> gate GEMM -> blend -> LN
// Block = 256 threads (8 waves), owns 32 rows (4 tokens) x 512 cols.
// Dynamic LDS: Abuf bf16[32][1032] (x | coupled concat), Cpl f32[32][516], Cs/amp.
// ---------------------------------------------------------------------------
#define APITCH 1032
#define CPITCH 516
#define SMEM_BYTES (32*APITCH*2 + 32*CPITCH*4 + 72*4)

__global__ __launch_bounds__(256, 2)
void fused_main(const float* __restrict__ x,
                const bf16_t* __restrict__ WcB,
                const bf16_t* __restrict__ WgB,
                const float* __restrict__ CsG,
                const float* __restrict__ ampG,
                const float* __restrict__ bg,
                const float* __restrict__ ln_w,
                const float* __restrict__ ln_b,
                float* __restrict__ out,
                float* __restrict__ mags) {
  extern __shared__ char smem[];
  bf16_t* Abuf = (bf16_t*)smem;                                  // [32][1032] bf16
  float*  Cpl  = (float*)(smem + 32*APITCH*2);                   // [32][516] f32
  float*  Cs   = (float*)(smem + 32*APITCH*2 + 32*CPITCH*4);     // 64
  float*  ampS = Cs + 64;                                        // 8

  const int tid  = threadIdx.x;
  const int w    = tid >> 5;
  const int lane = tid & 31;
  const int half = lane >> 4;
  const int mr   = lane & 15;
  const int nBase = w * 64;                 // wave's 64 output columns
  const int rowBase = blockIdx.x * 32;      // global row base (rows = (b,s,l) flat)

  // ---- Phase 0: stage x tile as bf16 into Abuf cols [0,512); copy C, amp ----
  {
    const float* xt = x + (size_t)rowBase * DD;   // 32*512 contiguous floats
    for (int i = tid; i < (32*512)/4; i += 256) {
      float4 v = ((const float4*)xt)[i];
      int flat = i * 4;
      int r = flat >> 9, k = flat & 511;
      bf16_t* dst = &Abuf[r*APITCH + k];
      dst[0] = (bf16_t)v.x; dst[1] = (bf16_t)v.y;
      dst[2] = (bf16_t)v.z; dst[3] = (bf16_t)v.w;
    }
    if (tid < 64) Cs[tid]  = CsG[tid];
    if (tid < 8)  ampS[tid] = ampG[tid];
  }
  __syncthreads();

  // ---- Phase 1: GEMM1 proj = x @ Wc^T  (per wave: 2 row tiles x 4 col tiles) ----
  v8f acc[2][4];
  {
    const v8f z = {0,0,0,0,0,0,0,0};
    #pragma unroll
    for (int rt = 0; rt < 2; ++rt)
      #pragma unroll
      for (int t = 0; t < 4; ++t) acc[rt][t] = z;

    for (int kk = 0; kk < 512; kk += 32) {
      FragBF a[2];
      #pragma unroll
      for (int rt = 0; rt < 2; ++rt) {
        const bf16_t* ap = &Abuf[(rt*16 + mr)*APITCH + kk + half*8];
        a[rt].q[0] = *(const uint4*)ap;          // K base..base+7
        a[rt].q[1] = *(const uint4*)(ap + 16);   // K base+16..base+23
      }
      #pragma unroll
      for (int t = 0; t < 4; ++t) {
        const int n = nBase + t*16 + mr;
        const bf16_t* bp = WcB + (size_t)n*512 + kk + half*8;  // B[k][n] = Wc[n][k]
        FragBF b;
        b.q[0] = *(const uint4*)bp;
        b.q[1] = *(const uint4*)(bp + 16);
        acc[0][t] = wmma_bf16(a[0].v, b.v, acc[0][t]);
        acc[1][t] = wmma_bf16(a[1].v, b.v, acc[1][t]);
      }
    }
  }

  // ---- Phase 2: coupling mix + ring exchange (per-lane: 8 VGPRs = 8 limbs) ----
  #pragma unroll
  for (int rt = 0; rt < 2; ++rt) {
    #pragma unroll
    for (int t = 0; t < 4; ++t) {
      float p[8], cp[8], ex[8];
      #pragma unroll
      for (int v = 0; v < 8; ++v) p[v] = acc[rt][t][v];
      #pragma unroll
      for (int i = 0; i < 8; ++i) {
        float s = 0.f;
        #pragma unroll
        for (int j = 0; j < 8; ++j) s += Cs[i*8 + j] * p[j];
        cp[i] = s;
      }
      #pragma unroll
      for (int i = 0; i < 8; ++i) ex[i] = ampS[i] * (p[(i+1)&7] - p[i]);
      #pragma unroll
      for (int i = 0; i < 8; ++i) cp[i] += 0.1f * (ex[i] - ex[(i+7)&7]);
      const int r0 = rt*16 + half*8;              // limb v -> tile row r0+v
      const int n  = nBase + t*16 + mr;
      #pragma unroll
      for (int v = 0; v < 8; ++v) {
        Cpl[(r0+v)*CPITCH + n] = cp[v];                    // f32 for blend
        Abuf[(r0+v)*APITCH + 512 + n] = (bf16_t)cp[v];     // concat for gate GEMM
      }
    }
  }
  __syncthreads();

  // ---- Phase 3: GEMM2 gatelogit = [x,coupled] @ Wg^T (K = 1024) ----
  v8f g[2][4];
  {
    const v8f z = {0,0,0,0,0,0,0,0};
    #pragma unroll
    for (int rt = 0; rt < 2; ++rt)
      #pragma unroll
      for (int t = 0; t < 4; ++t) g[rt][t] = z;

    for (int kk = 0; kk < 1024; kk += 32) {
      FragBF a[2];
      #pragma unroll
      for (int rt = 0; rt < 2; ++rt) {
        const bf16_t* ap = &Abuf[(rt*16 + mr)*APITCH + kk + half*8];
        a[rt].q[0] = *(const uint4*)ap;
        a[rt].q[1] = *(const uint4*)(ap + 16);
      }
      #pragma unroll
      for (int t = 0; t < 4; ++t) {
        const int n = nBase + t*16 + mr;
        const bf16_t* bp = WgB + (size_t)n*1024 + kk + half*8; // B[k][n] = Wg[n][k]
        FragBF b;
        b.q[0] = *(const uint4*)bp;
        b.q[1] = *(const uint4*)(bp + 16);
        g[0][t] = wmma_bf16(a[0].v, b.v, g[0][t]);
        g[1][t] = wmma_bf16(a[1].v, b.v, g[1][t]);
      }
    }
  }

  // ---- Phase 4: sigmoid gate + residual blend (in-place into Cpl) ----
  #pragma unroll
  for (int rt = 0; rt < 2; ++rt) {
    #pragma unroll
    for (int t = 0; t < 4; ++t) {
      const int n = nBase + t*16 + mr;
      const float bgn = bg[n];
      #pragma unroll
      for (int v = 0; v < 8; ++v) {
        const int tr = rt*16 + half*8 + v;
        float gate = 1.f / (1.f + expf(-(g[rt][t][v] + bgn)));
        float xv = x[(size_t)(rowBase + tr)*DD + n];
        float cv = Cpl[tr*CPITCH + n];
        Cpl[tr*CPITCH + n] = gate*cv + (1.f - gate)*xv;
      }
    }
  }
  __syncthreads();

  // ---- Phase 5: LayerNorm per row + store out + row magnitudes ----
  for (int rr = 0; rr < 4; ++rr) {
    const int tr = w*4 + rr;
    float vals[16];
    float s = 0.f, s2 = 0.f;
    #pragma unroll
    for (int i = 0; i < 16; ++i) {
      float v = Cpl[tr*CPITCH + lane + 32*i];
      vals[i] = v; s += v; s2 += v*v;
    }
    #pragma unroll
    for (int off = 16; off > 0; off >>= 1) {
      s  += __shfl_xor(s,  off, 32);
      s2 += __shfl_xor(s2, off, 32);
    }
    const float mu = s * (1.f/512.f);
    const float var = s2 * (1.f/512.f) - mu*mu;       // biased (torch default)
    const float rstd = rsqrtf(var + 1e-5f);
    const size_t gRow = (size_t)(rowBase + tr);
    float msq = 0.f;
    #pragma unroll
    for (int i = 0; i < 16; ++i) {
      const int col = lane + 32*i;
      float o = (vals[i] - mu) * rstd * ln_w[col] + ln_b[col];
      out[gRow*DD + col] = o;
      msq += o*o;
    }
    #pragma unroll
    for (int off = 16; off > 0; off >>= 1) msq += __shfl_xor(msq, off, 32);
    if (lane == 0) mags[gRow] = sqrtf(msq);
  }
}

// ---------------------------------------------------------------------------
// Loss reductions (deterministic fixed-order, '=' final writes)
// ---------------------------------------------------------------------------
// sm[b,l,d] = mean_s out[b,s,l,d]
__global__ void k_sm(const float* __restrict__ out, float* __restrict__ sm) {
  int i = blockIdx.x * 256 + threadIdx.x;  // 8*8*512 = 32768
  int d = i & 511, l = (i >> 9) & 7, b = i >> 12;
  float s = 0.f;
  for (int sidx = 0; sidx < SS; ++sidx)
    s += out[(((size_t)(b*SS + sidx))*LLIMB + l)*DD + d];
  sm[i] = s * (1.f/SS);
}

__global__ void k_energy(const float* __restrict__ sm, const float* __restrict__ omega,
                         const float* __restrict__ zp, float* __restrict__ dst) {
  __shared__ float red[64];
  int i = threadIdx.x;            // 64 = (b,l)
  int l = i & 7;
  float occ = 0.f;
  for (int d = 0; d < DD; ++d) { float v = sm[i*DD + d]; occ += v*v; }
  red[i] = omega[l]*(occ + 0.5f) + zp[0];
  __syncthreads();
  if (i == 0) {
    float s = 0.f;
    for (int j = 0; j < 64; ++j) s += red[j];
    s *= (1.f/BB);                                 // mean over b of per-b sums over l
    dst[0] = fabsf(s - (float)LLIMB * zp[0]) * 1e-4f;
  }
}

// mean over N=B*S per (l,d)
__global__ void k_mean(const float* __restrict__ out, float* __restrict__ meanLD) {
  int i = blockIdx.x * 256 + threadIdx.x;  // 4096 = (l,d)
  int l = i >> 9, d = i & 511;
  float s = 0.f;
  for (int n = 0; n < NTOK; ++n) s += out[((size_t)n*LLIMB + l)*DD + d];
  meanLD[i] = s * (1.f/NTOK);
}

__global__ void k_mom(const float* __restrict__ out, const float* __restrict__ meanLD,
                      float* __restrict__ m2, float* __restrict__ m3, float* __restrict__ m4) {
  int i = blockIdx.x * 256 + threadIdx.x;  // 4096
  int l = i >> 9, d = i & 511;
  const float mu = meanLD[i];
  float s2 = 0.f, s3 = 0.f, s4 = 0.f;
  for (int n = 0; n < NTOK; ++n) {
    float v = out[((size_t)n*LLIMB + l)*DD + d] - mu;
    float v2 = v*v;
    s2 += v2; s3 += v2*v; s4 += v2*v2;
  }
  m2[i] = s2; m3[i] = s3; m4[i] = s4;
}

__global__ void k_coh_final(const float* __restrict__ m2, const float* __restrict__ m3,
                            const float* __restrict__ m4, float* __restrict__ dst) {
  __shared__ float red[256];
  int tid = threadIdx.x;
  float part = 0.f;
  for (int i = tid; i < 4096; i += 256) {
    float stdv = sqrtf(m2[i] / (float)(NTOK - 1));   // unbiased (torch.std)
    float s3 = stdv*stdv*stdv;
    float sk = (m3[i] * (1.f/NTOK)) / (s3 + 1e-8f);
    float ku = (m4[i] * (1.f/NTOK)) / (s3*stdv + 1e-8f) - 3.f;
    part += sk*sk + ku*ku;
  }
  red[tid] = part;
  __syncthreads();
  for (int off = 128; off > 0; off >>= 1) {
    if (tid < off) red[tid] += red[tid + off];
    __syncthreads();
  }
  if (tid == 0) dst[0] = red[0] * (1.f/4096.f) * 0.01f;
}

__global__ void k_minmax(const float* __restrict__ mags,
                         float* __restrict__ mnA, float* __restrict__ mxA) {
  __shared__ float rmn[256], rmx[256];
  const int l = blockIdx.x, tid = threadIdx.x;
  float mn = 3.402823466e38f, mx = -3.402823466e38f;
  for (int t = tid; t < NTOK; t += 256) {
    float v = mags[(size_t)t*LLIMB + l];
    mn = fminf(mn, v); mx = fmaxf(mx, v);
  }
  rmn[tid] = mn; rmx[tid] = mx;
  __syncthreads();
  for (int off = 128; off > 0; off >>= 1) {
    if (tid < off) { rmn[tid] = fminf(rmn[tid], rmn[tid+off]);
                     rmx[tid] = fmaxf(rmx[tid], rmx[tid+off]); }
    __syncthreads();
  }
  if (tid == 0) { mnA[l] = rmn[0]; mxA[l] = rmx[0]; }
}

__global__ void k_quant(const float* __restrict__ mags, const float* __restrict__ mnA,
                        const float* __restrict__ mxA, float* __restrict__ qpart) {
  __shared__ float red[256];
  const int tid = threadIdx.x;
  float part = 0.f;
  for (int j = 0; j < 4; ++j) {
    int e = blockIdx.x * 1024 + j*256 + tid;   // 64 blocks x 1024 = 65536
    int l = e & 7;
    float mn = mnA[l], span = mxA[l] - mn;
    float m = mags[e];
    float best = 3.402823466e38f;
    #pragma unroll
    for (int k = 0; k < 8; ++k) {
      float lev = mn + span * ((float)k * (1.f/7.f));
      best = fminf(best, fabsf(m - lev));
    }
    part += best;
  }
  red[tid] = part;
  __syncthreads();
  for (int off = 128; off > 0; off >>= 1) {
    if (tid < off) red[tid] += red[tid + off];
    __syncthreads();
  }
  if (tid == 0) qpart[blockIdx.x] = red[0];
}

__global__ void k_quant_final(const float* __restrict__ qpart, float* __restrict__ dst) {
  __shared__ float red[64];
  int tid = threadIdx.x;
  red[tid] = qpart[tid];
  __syncthreads();
  if (tid == 0) {
    float s = 0.f;
    for (int j = 0; j < 64; ++j) s += red[j];
    dst[0] = s * (1.f/(float)NROW) * 0.001f;
  }
}

// ---------------------------------------------------------------------------
// Launcher
// ---------------------------------------------------------------------------
extern "C" void kernel_launch(void* const* d_in, const int* in_sizes, int n_in,
                              void* d_out, int out_size, void* d_ws, size_t ws_size,
                              hipStream_t stream) {
  const float* x        = (const float*)d_in[0];
  const float* coupling = (const float*)d_in[1];
  const float* omega    = (const float*)d_in[2];
  const float* zp       = (const float*)d_in[3];
  const float* Wc       = (const float*)d_in[4];
  const float* Wg       = (const float*)d_in[5];
  const float* bg       = (const float*)d_in[6];
  const float* ln_w     = (const float*)d_in[7];
  const float* ln_b     = (const float*)d_in[8];
  float* out = (float*)d_out;

  char* ws = (char*)d_ws;
  bf16_t* WcB   = (bf16_t*)(ws + 0);        // 512 KB
  bf16_t* WgB   = (bf16_t*)(ws + 524288);   // 1 MB
  float* Cs     = (float*)(ws + 1572864);   // 64 f32
  float* ampS   = (float*)(ws + 1573120);   // 8 f32
  float* sm     = (float*)(ws + 1573152);   // 32768 f32
  float* meanLD = (float*)(ws + 1704224);   // 4096 f32
  float* m2     = (float*)(ws + 1720608);   // 4096 f32
  float* m3     = (float*)(ws + 1736992);   // 4096 f32
  float* m4     = (float*)(ws + 1753376);   // 4096 f32
  float* mnA    = (float*)(ws + 1769760);   // 8 f32
  float* mxA    = (float*)(ws + 1769792);   // 8 f32
  float* qpart  = (float*)(ws + 1769824);   // 64 f32
  float* mags   = (float*)(ws + 1770080);   // 65536 f32   (total ~1.94 MB)

  hipFuncSetAttribute((const void*)fused_main,
                      hipFuncAttributeMaxDynamicSharedMemorySize, SMEM_BYTES);

  prep_small <<<1,    64,  0, stream>>>(coupling, Cs, ampS);
  cvt_weights<<<2048, 256, 0, stream>>>(Wc, Wg, WcB, WgB);
  fused_main <<<NROW/32, 256, SMEM_BYTES, stream>>>(x, WcB, WgB, Cs, ampS, bg,
                                                    ln_w, ln_b, out, mags);
  k_sm       <<<128, 256, 0, stream>>>(out, sm);
  k_energy   <<<1,   64,  0, stream>>>(sm, omega, zp, out + NOUT + 2);
  k_mean     <<<16,  256, 0, stream>>>(out, meanLD);
  k_mom      <<<16,  256, 0, stream>>>(out, meanLD, m2, m3, m4);
  k_coh_final<<<1,   256, 0, stream>>>(m2, m3, m4, out + NOUT);
  k_minmax   <<<8,   256, 0, stream>>>(mags, mnA, mxA);
  k_quant    <<<64,  256, 0, stream>>>(mags, mnA, mxA, qpart);
  k_quant_final<<<1, 64,  0, stream>>>(qpart, out + NOUT + 1);
}
